// PCLSurv_5617817223811
// MI455X (gfx1250) — compile-verified, matrix operand order checked by reading, not run on hardware
//
#include <hip/hip_runtime.h>

typedef __attribute__((ext_vector_type(16))) _Float16 v16h;
typedef __attribute__((ext_vector_type(8)))  _Float16 v8h;
typedef __attribute__((ext_vector_type(4)))  _Float16 v4h;
typedef __attribute__((ext_vector_type(8)))  float    v8f;
typedef __attribute__((ext_vector_type(4)))  float    v4f;
typedef __attribute__((ext_vector_type(2)))  float    v2f;

#define BT 64               // block tile (M and N)
#define KT 32               // K step per WMMA
#define LDS_K 40            // LDS row stride in halves: 80B = 16B-aligned, 20 banks
#define GEMM_THREADS 128    // 4 waves (wave32), each wave owns a 32x32 quadrant

// Lane's 16-half fragment from a [row][K] LDS tile: two contiguous 16B chunks.
// Per ISA 16-bit A layout: lane l (m=l&15, g=l>>4) holds K={g*8..+7, 16+g*8..+7}.
__device__ __forceinline__ v16h frag_from_lds(const _Float16* rowBase, int kHalf) {
    const _Float16* p = rowBase + kHalf;
    v8h lo = *(const v8h*)(p);
    v8h hi = *(const v8h*)(p + 16);
    return __builtin_shufflevector(lo, hi, 0,1,2,3,4,5,6,7,8,9,10,11,12,13,14,15);
}

__device__ __forceinline__ v8f wmma_step(v16h a, v16h b, v8f c) {
    return __builtin_amdgcn_wmma_f32_16x16x32_f16(false, a, false, b, (short)0, c, false, false);
}

// Move 8 consecutive fp32 of one row into dst[0..7] as fp16 (8B stores).
__device__ __forceinline__ void stage_row8(const float* src, _Float16* dst) {
    v2f x0 = *(const v2f*)(src);
    v2f x1 = *(const v2f*)(src + 2);
    v2f x2 = *(const v2f*)(src + 4);
    v2f x3 = *(const v2f*)(src + 6);
    v4f a = {x0.x, x0.y, x1.x, x1.y};
    v4f b = {x2.x, x2.y, x3.x, x3.y};
    *(v4h*)(dst)     = __builtin_convertvector(a, v4h);
    *(v4h*)(dst + 4) = __builtin_convertvector(b, v4h);
}

// ---------------------------------------------------------------------------
// C = act( A[MxK] * B[KxN] + bias[N] )  fp32 in, fp16 WMMA, fp32 accumulate
// act: 0=none 1=relu 2=sigmoid
// ---------------------------------------------------------------------------
__global__ __launch_bounds__(GEMM_THREADS)
void gemm_wmma_nn(const float* __restrict__ A, const float* __restrict__ B,
                  const float* __restrict__ bias, float* __restrict__ C,
                  int M, int N, int Kd, int act)
{
    __shared__ __align__(16) _Float16 As[BT][LDS_K];   // [m][k]
    __shared__ __align__(16) _Float16 Bs[BT][LDS_K];   // [n][k] (transposed tile)

    const int tid   = threadIdx.x;
    const int lane  = tid & 31;
    const int wave  = tid >> 5;
    const int warpM = wave & 1;
    const int warpN = wave >> 1;
    const int tileM = blockIdx.x * BT;
    const int tileN = blockIdx.y * BT;

    const int fRow  = lane & 15;
    const int kHalf = (lane >> 4) << 3;

    const bool fullM = (tileM + BT <= M);
    const bool fullN = (tileN + BT <= N);

    const int aRow = tid >> 1, aK = (tid & 1) << 4;   // A: 64 rows x 32 k, 16 elems/thread
    const int bK   = tid >> 2, bN = (tid & 3) << 4;   // B: 32 k x 64 n, 16 elems/thread

    v8f acc[2][2] = {};

    const int kEnd = Kd & ~(KT - 1);
    int k0 = 0;
    for (; k0 < kEnd; k0 += KT) {                      // guard-free K body
        if (fullM) {
            const float* s = A + (tileM + aRow) * Kd + k0 + aK;
            stage_row8(s,     &As[aRow][aK]);
            stage_row8(s + 8, &As[aRow][aK + 8]);
        } else {
            for (int i = tid; i < BT * KT; i += GEMM_THREADS) {
                int r = i >> 5, c = i & 31;
                int gr = tileM + r;
                As[r][c] = (_Float16)((gr < M) ? A[gr * Kd + k0 + c] : 0.f);
            }
        }
        if (fullN) {
            const float* s = B + (k0 + bK) * N + tileN + bN;
            #pragma unroll
            for (int j = 0; j < 16; j += 2) {
                v2f x = *(const v2f*)(s + j);
                Bs[bN + j][bK]     = (_Float16)x.x;
                Bs[bN + j + 1][bK] = (_Float16)x.y;
            }
        } else {
            for (int i = tid; i < KT * BT; i += GEMM_THREADS) {
                int k = i >> 6, n = i & 63;
                int gn = tileN + n;
                Bs[n][k] = (_Float16)((gn < N) ? B[(k0 + k) * N + gn] : 0.f);
            }
        }
        __syncthreads();

        if (k0 + KT < Kd)
            __builtin_prefetch(A + (tileM + aRow) * Kd + k0 + KT + aK, 0, 1);

        v16h af0 = frag_from_lds(&As[warpM * 32 + fRow][0], kHalf);
        v16h af1 = frag_from_lds(&As[warpM * 32 + 16 + fRow][0], kHalf);
        v16h bf0 = frag_from_lds(&Bs[warpN * 32 + fRow][0], kHalf);
        v16h bf1 = frag_from_lds(&Bs[warpN * 32 + 16 + fRow][0], kHalf);
        acc[0][0] = wmma_step(af0, bf0, acc[0][0]);
        acc[0][1] = wmma_step(af0, bf1, acc[0][1]);
        acc[1][0] = wmma_step(af1, bf0, acc[1][0]);
        acc[1][1] = wmma_step(af1, bf1, acc[1][1]);
        __syncthreads();
    }
    if (k0 < Kd) {                                     // guarded K tail
        for (int i = tid; i < BT * KT; i += GEMM_THREADS) {
            int r = i >> 5, c = i & 31;
            int gr = tileM + r, gk = k0 + c;
            As[r][c] = (_Float16)((gr < M && gk < Kd) ? A[gr * Kd + gk] : 0.f);
        }
        for (int i = tid; i < KT * BT; i += GEMM_THREADS) {
            int k = i >> 6, n = i & 63;
            int gn = tileN + n, gk = k0 + k;
            Bs[n][k] = (_Float16)((gn < N && gk < Kd) ? B[gk * N + gn] : 0.f);
        }
        __syncthreads();
        v16h af0 = frag_from_lds(&As[warpM * 32 + fRow][0], kHalf);
        v16h af1 = frag_from_lds(&As[warpM * 32 + 16 + fRow][0], kHalf);
        v16h bf0 = frag_from_lds(&Bs[warpN * 32 + fRow][0], kHalf);
        v16h bf1 = frag_from_lds(&Bs[warpN * 32 + 16 + fRow][0], kHalf);
        acc[0][0] = wmma_step(af0, bf0, acc[0][0]);
        acc[0][1] = wmma_step(af0, bf1, acc[0][1]);
        acc[1][0] = wmma_step(af1, bf0, acc[1][0]);
        acc[1][1] = wmma_step(af1, bf1, acc[1][1]);
    }

    const int rOff = (lane >> 4) << 3;
    const int nCol = lane & 15;
    #pragma unroll
    for (int mi = 0; mi < 2; ++mi) {
        #pragma unroll
        for (int ni = 0; ni < 2; ++ni) {
            int col = tileN + warpN * 32 + ni * 16 + nCol;
            if (col >= N) continue;
            float bsv = bias ? bias[col] : 0.f;
            int rowB = tileM + warpM * 32 + mi * 16 + rOff;
            #pragma unroll
            for (int r = 0; r < 8; ++r) {
                int row = rowB + r;
                if (row < M) {
                    float v = acc[mi][ni][r] + bsv;
                    if (act == 1)      v = fmaxf(v, 0.f);
                    else if (act == 2) v = 1.f / (1.f + __expf(-v));
                    C[row * N + col] = v;
                }
            }
        }
    }
}

// ---------------------------------------------------------------------------
// C = A[MxK] * Bt[NxK]^T  (NT GEMM).  mode 0: plain store (ldc given).
// mode 1/2: off-diagonal epilogue into logits[ldc=4095]:
//   skip j==row, outcol = base + (j<row ? j : j-1), base = 1 (mode1) / 2048 (mode2)
// ---------------------------------------------------------------------------
__global__ __launch_bounds__(GEMM_THREADS)
void gemm_wmma_nt(const float* __restrict__ A, const float* __restrict__ Bt,
                  float* __restrict__ C, int M, int N, int Kd, int ldc,
                  int mode, float scale)
{
    __shared__ __align__(16) _Float16 As[BT][LDS_K];   // [m][k]
    __shared__ __align__(16) _Float16 Bs[BT][LDS_K];   // [n][k]

    const int tid   = threadIdx.x;
    const int lane  = tid & 31;
    const int wave  = tid >> 5;
    const int warpM = wave & 1;
    const int warpN = wave >> 1;
    const int tileM = blockIdx.x * BT;
    const int tileN = blockIdx.y * BT;

    const int fRow  = lane & 15;
    const int kHalf = (lane >> 4) << 3;

    const bool fullM = (tileM + BT <= M);
    const bool fullN = (tileN + BT <= N);

    const int aRow = tid >> 1, aK = (tid & 1) << 4;

    v8f acc[2][2] = {};

    const int kEnd = Kd & ~(KT - 1);
    int k0 = 0;
    for (; k0 < kEnd; k0 += KT) {
        if (fullM) {
            const float* s = A + (tileM + aRow) * Kd + k0 + aK;
            stage_row8(s,     &As[aRow][aK]);
            stage_row8(s + 8, &As[aRow][aK + 8]);
        } else {
            for (int i = tid; i < BT * KT; i += GEMM_THREADS) {
                int r = i >> 5, c = i & 31;
                int gr = tileM + r;
                As[r][c] = (_Float16)((gr < M) ? A[gr * Kd + k0 + c] : 0.f);
            }
        }
        if (fullN) {                                   // Bt rows already [n][k]
            const float* s = Bt + (tileN + aRow) * Kd + k0 + aK;
            stage_row8(s,     &Bs[aRow][aK]);
            stage_row8(s + 8, &Bs[aRow][aK + 8]);
        } else {
            for (int i = tid; i < BT * KT; i += GEMM_THREADS) {
                int n = i >> 5, k = i & 31;
                int gn = tileN + n;
                Bs[n][k] = (_Float16)((gn < N) ? Bt[gn * Kd + k0 + k] : 0.f);
            }
        }
        __syncthreads();

        v16h af0 = frag_from_lds(&As[warpM * 32 + fRow][0], kHalf);
        v16h af1 = frag_from_lds(&As[warpM * 32 + 16 + fRow][0], kHalf);
        v16h bf0 = frag_from_lds(&Bs[warpN * 32 + fRow][0], kHalf);
        v16h bf1 = frag_from_lds(&Bs[warpN * 32 + 16 + fRow][0], kHalf);
        acc[0][0] = wmma_step(af0, bf0, acc[0][0]);
        acc[0][1] = wmma_step(af0, bf1, acc[0][1]);
        acc[1][0] = wmma_step(af1, bf0, acc[1][0]);
        acc[1][1] = wmma_step(af1, bf1, acc[1][1]);
        __syncthreads();
    }
    if (k0 < Kd) {
        for (int i = tid; i < BT * KT; i += GEMM_THREADS) {
            int r = i >> 5, c = i & 31;
            int gr = tileM + r, gk = k0 + c;
            As[r][c] = (_Float16)((gr < M && gk < Kd) ? A[gr * Kd + gk] : 0.f);
        }
        for (int i = tid; i < BT * KT; i += GEMM_THREADS) {
            int n = i >> 5, k = i & 31;
            int gn = tileN + n, gk = k0 + k;
            Bs[n][k] = (_Float16)((gn < N && gk < Kd) ? Bt[gn * Kd + gk] : 0.f);
        }
        __syncthreads();
        v16h af0 = frag_from_lds(&As[warpM * 32 + fRow][0], kHalf);
        v16h af1 = frag_from_lds(&As[warpM * 32 + 16 + fRow][0], kHalf);
        v16h bf0 = frag_from_lds(&Bs[warpN * 32 + fRow][0], kHalf);
        v16h bf1 = frag_from_lds(&Bs[warpN * 32 + 16 + fRow][0], kHalf);
        acc[0][0] = wmma_step(af0, bf0, acc[0][0]);
        acc[0][1] = wmma_step(af0, bf1, acc[0][1]);
        acc[1][0] = wmma_step(af1, bf0, acc[1][0]);
        acc[1][1] = wmma_step(af1, bf1, acc[1][1]);
    }

    const int rOff = (lane >> 4) << 3;
    const int nCol = lane & 15;
    #pragma unroll
    for (int mi = 0; mi < 2; ++mi) {
        #pragma unroll
        for (int ni = 0; ni < 2; ++ni) {
            int col = tileN + warpN * 32 + ni * 16 + nCol;
            if (col >= N) continue;
            int rowB = tileM + warpM * 32 + mi * 16 + rOff;
            #pragma unroll
            for (int r = 0; r < 8; ++r) {
                int row = rowB + r;
                if (row >= M) continue;
                float v = acc[mi][ni][r] * scale;
                if (mode == 0) {
                    C[row * ldc + col] = v;
                } else {
                    if (col == row) continue;            // drop diagonal
                    int base   = (mode == 1) ? 1 : 2048;
                    int outcol = base + ((col < row) ? col : (col - 1));
                    C[row * ldc + outcol] = v;
                }
            }
        }
    }
}

// ---------------------------------------------------------------------------
// Per-row fusion: s_n, fbn -> X[2048x60]; l_pos -> logits[:,0]; labels = 0
// ---------------------------------------------------------------------------
__global__ __launch_bounds__(64)
void row_misc_kernel(const float* __restrict__ q, const float* __restrict__ k,
                     const float* __restrict__ qlr, const float* __restrict__ klm,
                     float* __restrict__ X, float* __restrict__ logits,
                     float* __restrict__ labels, float* __restrict__ labels_proto)
{
    const int n   = blockIdx.x;
    const int tid = threadIdx.x;
    __shared__ float red1[64], red2[64];

    float sv = 0.f, ss = 0.f, pd = 0.f;
    if (tid < 50) {
        float qv = q[n * 50 + tid];
        float kv = k[n * 50 + tid];
        sv = qv + kv;
        ss = sv * sv;
        pd = qv * kv;
    }
    red1[tid] = ss; red2[tid] = pd;
    __syncthreads();
    for (int s = 32; s > 0; s >>= 1) {
        if (tid < s) { red1[tid] += red1[tid + s]; red2[tid] += red2[tid + s]; }
        __syncthreads();
    }

    float inv = 1.f / fmaxf(sqrtf(red1[0]), 1e-12f);
    if (tid < 50) X[n * 60 + tid] = sv * inv;
    if (tid < 10) {
        float t = 0.f;
        #pragma unroll
        for (int i = 0; i < 5; ++i)
            t += qlr[n * 50 + i * 10 + tid] * klm[n * 50 + i * 10 + tid];
        X[n * 60 + 50 + tid] = sqrtf(fmaxf(t, 0.f)) - sqrtf(fmaxf(-t, 0.f));
    }
    if (tid == 0) {
        logits[n * 4095] = red2[0] * 5.0f;   // l_pos / T
        labels[n] = 0.f;
        labels_proto[n] = 0.f;
    }
}

// ---------------------------------------------------------------------------
// logits_proto: [full[n,c] , full[n, neg_ids]]
// ---------------------------------------------------------------------------
__global__ __launch_bounds__(64)
void proto_gather_kernel(const float* __restrict__ full, const int* __restrict__ c2c,
                         float* __restrict__ out)
{
    const int n = blockIdx.x;
    const int j = threadIdx.x;   // 0..63
    const int c = c2c[n];
    float v;
    if (j == 0) v = full[n * 64 + c];
    else {
        int i = j - 1;
        int neg = i + ((i >= c) ? 1 : 0);
        v = full[n * 64 + neg];
    }
    out[n * 64 + j] = v;
}

// ---------------------------------------------------------------------------
extern "C" void kernel_launch(void* const* d_in, const int* in_sizes, int n_in,
                              void* d_out, int out_size, void* d_ws, size_t ws_size,
                              hipStream_t stream)
{
    const int N = 2048, D_R = 20000, D_M = 2000;

    const float* omics_q = (const float*)d_in[0];
    const float* omics_k = (const float*)d_in[1];
    const float* protos  = (const float*)d_in[2];
    const float* re_w1 = (const float*)d_in[3];  const float* re_b1 = (const float*)d_in[4];
    const float* re_w2 = (const float*)d_in[5];  const float* re_b2 = (const float*)d_in[6];
    const float* re_w3 = (const float*)d_in[7];  const float* re_b3 = (const float*)d_in[8];
    const float* rd_w1 = (const float*)d_in[9];  const float* rd_b1 = (const float*)d_in[10];
    const float* rd_w2 = (const float*)d_in[11]; const float* rd_b2 = (const float*)d_in[12];
    const float* rd_w3 = (const float*)d_in[13]; const float* rd_b3 = (const float*)d_in[14];
    const float* me_w1 = (const float*)d_in[15]; const float* me_b1 = (const float*)d_in[16];
    const float* me_w2 = (const float*)d_in[17]; const float* me_b2 = (const float*)d_in[18];
    const float* me_w3 = (const float*)d_in[19]; const float* me_b3 = (const float*)d_in[20];
    const float* md_w1 = (const float*)d_in[21]; const float* md_b1 = (const float*)d_in[22];
    const float* md_w2 = (const float*)d_in[23]; const float* md_b2 = (const float*)d_in[24];
    const float* md_w3 = (const float*)d_in[25]; const float* md_b3 = (const float*)d_in[26];
    const float* lr_w  = (const float*)d_in[27]; const float* lr_b  = (const float*)d_in[28];
    const float* lm_w  = (const float*)d_in[29]; const float* lm_b  = (const float*)d_in[30];
    const int*   c2c   = (const int*)d_in[31];

    float* out          = (float*)d_out;
    float* rna          = out;
    float* mirna        = rna   + (size_t)N * D_R;
    float* logits       = mirna + (size_t)N * D_M;
    float* labels       = logits + (size_t)N * 4095;
    float* logits_proto = labels + N;
    float* labels_proto = logits_proto + (size_t)N * 64;

    float* ws  = (float*)d_ws;
    float* q1  = ws; ws += (size_t)N * 200;
    float* k1  = ws; ws += (size_t)N * 200;
    float* qt  = ws; ws += (size_t)N * 50;
    float* q   = ws; ws += (size_t)N * 50;
    float* kt  = ws; ws += (size_t)N * 50;
    float* kk  = ws; ws += (size_t)N * 50;
    float* h1  = ws; ws += (size_t)N * 50;
    float* h2  = ws; ws += (size_t)N * 200;
    float* qlr = ws; ws += (size_t)N * 50;
    float* klm = ws; ws += (size_t)N * 50;
    float* X   = ws; ws += (size_t)N * 60;
    float* full= ws; ws += (size_t)N * 64;

    auto nn = [&](const float* A, const float* B, const float* bias, float* C,
                  int M, int Nn, int Kd, int act) {
        dim3 g((M + BT - 1) / BT, (Nn + BT - 1) / BT);
        hipLaunchKernelGGL(gemm_wmma_nn, g, dim3(GEMM_THREADS), 0, stream,
                           A, B, bias, C, M, Nn, Kd, act);
    };
    auto nt = [&](const float* A, const float* Bt, float* C,
                  int M, int Nn, int Kd, int ldc, int mode, float scale) {
        dim3 g((M + BT - 1) / BT, (Nn + BT - 1) / BT);
        hipLaunchKernelGGL(gemm_wmma_nt, g, dim3(GEMM_THREADS), 0, stream,
                           A, Bt, C, M, Nn, Kd, ldc, mode, scale);
    };

    // RNA encoder
    nn(omics_q, re_w1, re_b1, q1, N, 200, D_R, 1);
    nn(q1,      re_w2, re_b2, qt, N,  50, 200, 1);
    nn(qt,      re_w3, re_b3, q,  N,  50,  50, 0);
    // miRNA encoder
    nn(omics_k, me_w1, me_b1, k1, N, 200, D_M, 1);
    nn(k1,      me_w2, me_b2, kt, N,  50, 200, 1);
    nn(kt,      me_w3, me_b3, kk, N,  50,  50, 0);
    // RNA decoder -> rna_new (sigmoid)
    nn(q,  rd_w1, rd_b1, h1, N,  50,  50, 1);
    nn(h1, rd_w2, rd_b2, h2, N, 200,  50, 1);
    nn(h2, rd_w3, rd_b3, rna, N, D_R, 200, 2);
    // miRNA decoder -> mirna_new (sigmoid)
    nn(kk, md_w1, md_b1, h1, N,  50,  50, 1);
    nn(h1, md_w2, md_b2, h2, N, 200,  50, 1);
    nn(h2, md_w3, md_b3, mirna, N, D_M, 200, 2);
    // bilinear fusion inputs
    nn(q,  lr_w, lr_b, qlr, N, 50, 50, 0);
    nn(kk, lm_w, lm_b, klm, N, 50, 50, 0);

    // X, l_pos, labels
    hipLaunchKernelGGL(row_misc_kernel, dim3(N), dim3(64), 0, stream,
                       q, kk, qlr, klm, X, logits, labels, labels_proto);

    // contrastive logits (off-diag of q@q^T and q@k^T, scaled by 1/T)
    nt(q, q,  logits, N, N, 50, 4095, 1, 5.0f);
    nt(q, kk, logits, N, N, 50, 4095, 2, 5.0f);

    // prototype scores
    nt(X, protos, full, N, 64, 60, 64, 0, 1.0f);
    hipLaunchKernelGGL(proto_gather_kernel, dim3(N), dim3(64), 0, stream,
                       full, c2c, logits_proto);
}
